// TrellisLinear_38646115729896
// MI455X (gfx1250) — matrix operand order repaired and checked by reading, not compiled, not run on hardware
//
#include <hip/hip_runtime.h>

// ---------------------------------------------------------------------------
// TrellisLinear for MI455X (gfx1250): FWHT + u8 fake-quant + 2-bit trellis
// decode + FWHT/scale, then bf16 WMMA GEMM (v_wmma_f32_16x16x32_bf16),
// double-buffered LDS (single loop body, no unroll -> no acc shuffling).
// ---------------------------------------------------------------------------

#define M_ROWS 8192      // 4 * 2048
#define N_OUT  4096
#define K_IN   4096
#define NTHREADS_FWHT 512

typedef __attribute__((ext_vector_type(16))) __bf16 v16bf;
typedef __attribute__((ext_vector_type(8)))  float  v8f;

union FragBF {
    uint4 u[2];
    v16bf v;
};

union Pack8 {
    unsigned short s[8];
    uint4 u;
};

static __device__ inline unsigned short f2bf(float f) {
    unsigned int x = __float_as_uint(f);
    unsigned int r = (x + 0x7FFFu + ((x >> 16) & 1u)) >> 16;   // RNE
    return (unsigned short)r;
}

// In-LDS FWHT over 4096 elements, NTHREADS_FWHT threads (no normalization).
static __device__ inline void fwht4096(float* s, int tid) {
    for (int h = 1; h < 4096; h <<= 1) {
        __syncthreads();
        #pragma unroll
        for (int p = tid; p < 2048; p += NTHREADS_FWHT) {
            int i = ((p & ~(h - 1)) << 1) | (p & (h - 1));
            float a = s[i];
            float b = s[i + h];
            s[i]     = a + b;
            s[i + h] = a - b;
        }
    }
    __syncthreads();
}

// ---------------------------------------------------------------------------
// K1: xh[row] = FWHT(x[row] * SU) * (1/64)
// ---------------------------------------------------------------------------
__global__ void __launch_bounds__(NTHREADS_FWHT)
k_rowfwht(const float* __restrict__ x, const float* __restrict__ SU,
          float* __restrict__ xh) {
    __shared__ float s[4096];
    const int row = blockIdx.x;
    const int tid = threadIdx.x;
    const float* xr = x + (size_t)row * K_IN;
    #pragma unroll
    for (int j = 0; j < 8; ++j) {
        int c = tid + j * NTHREADS_FWHT;
        s[c] = xr[c] * SU[c];
    }
    fwht4096(s, tid);
    float* xo = xh + (size_t)row * K_IN;
    #pragma unroll
    for (int j = 0; j < 8; ++j) {
        int c = tid + j * NTHREADS_FWHT;
        xo[c] = s[c] * (1.0f / 64.0f);
    }
}

// ---------------------------------------------------------------------------
// K2: per-channel min/max over all rows -> scale, zero. 32 channels / block.
// ---------------------------------------------------------------------------
__global__ void __launch_bounds__(256)
k_minmax(const float* __restrict__ xh, float* __restrict__ scale,
         float* __restrict__ zero) {
    __shared__ float smn[256], smx[256];
    const int tid   = threadIdx.x;
    const int c     = tid & 31;
    const int g     = tid >> 5;
    const int cbase = blockIdx.x * 32;
    float mn = 3.402823466e38f, mx = -3.402823466e38f;
    for (int r = g; r < M_ROWS; r += 8) {
        float v = xh[(size_t)r * K_IN + cbase + c];
        mn = fminf(mn, v);
        mx = fmaxf(mx, v);
    }
    smn[tid] = mn; smx[tid] = mx;
    __syncthreads();
    for (int s = 128; s >= 32; s >>= 1) {
        if (tid < s) {
            smn[tid] = fminf(smn[tid], smn[tid + s]);
            smx[tid] = fmaxf(smx[tid], smx[tid + s]);
        }
        __syncthreads();
    }
    if (tid < 32) {
        float lo = smn[tid], hi = smx[tid];
        float sc = fmaxf((hi - lo) * (1.0f / 255.0f), 1e-8f);
        scale[cbase + tid] = sc;
        zero[cbase + tid]  = rintf(-lo / sc);
    }
}

// ---------------------------------------------------------------------------
// K3: fake-quant xh and pack bf16 activations A (row-major M x K).
// ---------------------------------------------------------------------------
__global__ void __launch_bounds__(256)
k_quant(const float* __restrict__ xh, const float* __restrict__ scale,
        const float* __restrict__ zero, unsigned short* __restrict__ A) {
    const size_t base = ((size_t)blockIdx.x * 256 + threadIdx.x) * 4;
    const int c = (int)(base & (K_IN - 1));
    float4 v = *(const float4*)(xh + base);
    float out[4];
    const float* vp = (const float*)&v;
    #pragma unroll
    for (int j = 0; j < 4; ++j) {
        float sc = scale[c + j];
        float zp = zero[c + j];
        float q  = fminf(fmaxf(rintf(vp[j] / sc) + zp, 0.0f), 255.0f);
        out[j]   = (q - zp) * sc;
    }
    ushort4 pk;
    pk.x = f2bf(out[0]); pk.y = f2bf(out[1]);
    pk.z = f2bf(out[2]); pk.w = f2bf(out[3]);
    *(ushort4*)(A + base) = pk;
}

// ---------------------------------------------------------------------------
// K4: 8 adjacent input columns per block (they share the trellis word and
// the group-scale row). Decode for all o, FWHT over OUT per column,
// * SV[o] * group_scale, store W bf16 in natural (OUT x IN) row-major
// layout with 16B-contiguous stores. 128 KB dynamic LDS (WGP has 320 KB).
// ---------------------------------------------------------------------------
__global__ void __launch_bounds__(NTHREADS_FWHT)
k_wdecode(const int* __restrict__ trellis, const float* __restrict__ SV,
          const float* __restrict__ wscale, unsigned short* __restrict__ W) {
    extern __shared__ float s[];           // [8][4096]
    const int i0  = blockIdx.x * 8;        // first input channel of this block
    const int tid = threadIdx.x;
    const int widx  = (i0 >> 3) & 31;      // word within trellis row (shared)
    const int ihigh = i0 >> 8;             // group-scale column (shared)
    #pragma unroll
    for (int t = 0; t < 8; ++t) {
        int o = tid + t * NTHREADS_FWHT;
        int trow = o * 16 + ihigh;
        unsigned w = (unsigned)trellis[trow * 32 + widx];
        #pragma unroll
        for (int j = 0; j < 8; ++j) {
            int code = (int)((w >> (14 - 2 * j)) & 3u);
            s[j * 4096 + o] = (float)(2 * code - 3) * 0.44721359549995793f;
        }
    }
    // FWHT over OUT for the 8 columns, interleaved per stage.
    for (int h = 1; h < 4096; h <<= 1) {
        __syncthreads();
        for (int p = tid; p < 2048; p += NTHREADS_FWHT) {
            int ia = ((p & ~(h - 1)) << 1) | (p & (h - 1));
            #pragma unroll
            for (int j = 0; j < 8; ++j) {
                float* c = s + j * 4096;
                float a = c[ia];
                float b = c[ia + h];
                c[ia]     = a + b;
                c[ia + h] = a - b;
            }
        }
    }
    __syncthreads();
    #pragma unroll
    for (int t = 0; t < 8; ++t) {
        int o = tid + t * NTHREADS_FWHT;
        int trow = o * 16 + ihigh;
        float sc = (1.0f / 64.0f) * SV[o] * wscale[trow];
        Pack8 pk;
        #pragma unroll
        for (int j = 0; j < 8; ++j)
            pk.s[j] = f2bf(s[j * 4096 + o] * sc);
        *(uint4*)(W + (size_t)o * K_IN + i0) = pk.u;   // 16B contiguous
    }
}

// ---------------------------------------------------------------------------
// K5: GEMM  out[m][n] = sum_k A[m][k] * W[n][k]   (bf16 x bf16 -> f32)
// Block tile 128x128, K step 32, double-buffered LDS (register-staged).
// Single loop body (unroll disabled) so accumulators stay pinned.
// 8 waves of 32; wave tile 32x64 (2x4 WMMA tiles of 16x16x32).
// ---------------------------------------------------------------------------
__global__ void __launch_bounds__(256)
k_gemm(const unsigned short* __restrict__ A, const unsigned short* __restrict__ W,
       float* __restrict__ out) {
    __shared__ __align__(16) unsigned short Ash[2 * 128 * 32];  // [buf][m][k]
    __shared__ __align__(16) unsigned short Bsh[2 * 128 * 32];  // [buf][n][k]

    const int tid  = threadIdx.x;
    const int wave = tid >> 5;
    const int lane = tid & 31;
    const int n0   = blockIdx.x * 128;
    const int m0   = blockIdx.y * 128;
    const int wm   = wave >> 1;      // 0..3 -> 32-row slice
    const int wn   = wave & 1;       // 0..1 -> 64-col slice
    const int r    = lane & 15;
    const int kh   = lane >> 4;      // K-half select per WMMA layout

    const int srow = tid >> 1;       // staging: 128 rows x 2 halves
    const int shalf = tid & 1;
    const unsigned short* aptr = A + (size_t)(m0 + srow) * K_IN + shalf * 16;
    const unsigned short* bptr = W + (size_t)(n0 + srow) * K_IN + shalf * 16;
    const int soff = srow * 32 + shalf * 16;

    // fragment base offsets within a buffer (element units)
    const int aoff0 = (wm * 32 + r) * 32 + kh * 8;          // + am*16*32
    const int boff0 = (wn * 64 + r) * 32 + kh * 8;          // + bn*16*32

    v8f acc[2][4];
    #pragma unroll
    for (int am = 0; am < 2; ++am)
        #pragma unroll
        for (int bn = 0; bn < 4; ++bn)
            acc[am][bn] = (v8f){0.f,0.f,0.f,0.f,0.f,0.f,0.f,0.f};

    // ---- prologue: stage K-tile 0 into buffer 0 ----
    uint4 ra0 = *(const uint4*)(aptr);
    uint4 ra1 = *(const uint4*)(aptr + 8);
    uint4 rb0 = *(const uint4*)(bptr);
    uint4 rb1 = *(const uint4*)(bptr + 8);
    *(uint4*)&Ash[soff]     = ra0;
    *(uint4*)&Ash[soff + 8] = ra1;
    *(uint4*)&Bsh[soff]     = rb0;
    *(uint4*)&Bsh[soff + 8] = rb1;
    __syncthreads();

    const int NT = K_IN / 32;
    int bufoff = 0;                         // toggles 0 <-> 128*32
    #pragma clang loop unroll(disable)
    for (int kt = 0; kt < NT; ++kt) {
        // ---- issue next tile's global loads (latency overlaps WMMA) ----
        if (kt + 1 < NT) {
            const int k1 = (kt + 1) * 32;
            ra0 = *(const uint4*)(aptr + k1);
            ra1 = *(const uint4*)(aptr + k1 + 8);
            rb0 = *(const uint4*)(bptr + k1);
            rb1 = *(const uint4*)(bptr + k1 + 8);
            __builtin_prefetch(aptr + k1 + 32, 0, 0);   // global_prefetch_b8
            __builtin_prefetch(bptr + k1 + 32, 0, 0);
        }

        // ---- compute: per-wave 2x4 WMMA tiles from current buffer ----
        FragBF fa[2], fb[4];
        #pragma unroll
        for (int am = 0; am < 2; ++am) {
            const unsigned short* p = &Ash[bufoff + aoff0 + am * (16 * 32)];
            // lanes 0-15: K 0..7 | 16..23 ; lanes 16-31: K 8..15 | 24..31
            fa[am].u[0] = *(const uint4*)(p);
            fa[am].u[1] = *(const uint4*)(p + 16);
        }
        #pragma unroll
        for (int bn = 0; bn < 4; ++bn) {
            const unsigned short* p = &Bsh[bufoff + boff0 + bn * (16 * 32)];
            // lanes 0-15: K 0..7 | 16..23 ; lanes 16-31: K 8..15 | 24..31
            fb[bn].u[0] = *(const uint4*)(p);
            fb[bn].u[1] = *(const uint4*)(p + 16);
        }
        #pragma unroll
        for (int am = 0; am < 2; ++am)
            #pragma unroll
            for (int bn = 0; bn < 4; ++bn)
                acc[am][bn] = __builtin_amdgcn_wmma_f32_16x16x32_bf16(
                    false, fa[am].v, false, fb[bn].v,
                    (short)0, acc[am][bn], false, false);

        // ---- commit next tile to the alternate buffer ----
        if (kt + 1 < NT) {
            const int nxt = bufoff ^ (128 * 32);
            __syncthreads();
            *(uint4*)&Ash[nxt + soff]     = ra0;
            *(uint4*)&Ash[nxt + soff + 8] = ra1;
            *(uint4*)&Bsh[nxt + soff]     = rb0;
            *(uint4*)&Bsh[nxt + soff + 8] = rb1;
            __syncthreads();
            bufoff = nxt;
        }
    }

    // ---- epilogue: C layout VGPR v -> M = v + 8*(lane>>4), N = lane&15 ----
    #pragma unroll
    for (int am = 0; am < 2; ++am) {
        #pragma unroll
        for (int bn = 0; bn < 4; ++bn) {
            #pragma unroll
            for (int v = 0; v < 8; ++v) {
                int row = m0 + wm * 32 + am * 16 + kh * 8 + v;
                int col = n0 + wn * 64 + bn * 16 + r;
                out[(size_t)row * N_OUT + col] = acc[am][bn][v];
            }
        }
    }
}

// ---------------------------------------------------------------------------
// Launcher
// ---------------------------------------------------------------------------
extern "C" void kernel_launch(void* const* d_in, const int* in_sizes, int n_in,
                              void* d_out, int out_size, void* d_ws, size_t ws_size,
                              hipStream_t stream) {
    const float* x       = (const float*)d_in[0];
    const float* SU      = (const float*)d_in[1];
    const float* SV      = (const float*)d_in[2];
    const float* wscale  = (const float*)d_in[3];
    const int*   trellis = (const int*)d_in[4];
    float* out = (float*)d_out;

    char* ws = (char*)d_ws;
    float*          xh    = (float*)(ws);                                  // 128 MiB
    unsigned short* Abf   = (unsigned short*)(ws + (size_t)134217728);     //  64 MiB
    unsigned short* Wbf   = (unsigned short*)(ws + (size_t)201326592);     //  32 MiB (OUT x IN)
    float*          scale = (float*)(ws + (size_t)234881024);              //  16 KiB
    float*          zero  = (float*)(ws + (size_t)234897408);              //  16 KiB

    // allow 128 KB dynamic LDS for the decode kernel (WGP has 320 KB)
    (void)hipFuncSetAttribute((const void*)k_wdecode,
                              hipFuncAttributeMaxDynamicSharedMemorySize, 131072);

    k_rowfwht<<<M_ROWS, NTHREADS_FWHT, 0, stream>>>(x, SU, xh);
    k_minmax<<<K_IN / 32, 256, 0, stream>>>(xh, scale, zero);
    k_quant<<<(M_ROWS * (size_t)K_IN) / (256 * 4), 256, 0, stream>>>(xh, scale, zero, Abf);
    k_wdecode<<<K_IN / 8, NTHREADS_FWHT, 131072, stream>>>(trellis, SV, wscale, Wbf);
    k_gemm<<<dim3(N_OUT / 128, M_ROWS / 128), 256, 0, stream>>>(Abf, Wbf, out);

    (void)in_sizes; (void)n_in; (void)out_size; (void)ws_size;
}
